// TransformerEncoder_37082747633708
// MI455X (gfx1250) — matrix operand order repaired and checked by reading, not compiled, not run on hardware
//
#include <hip/hip_runtime.h>
#include <hip/hip_bf16.h>

// ---------------------------------------------------------------------------
// CDNA5 WMMA types (wave32): v_wmma_f32_16x16x32_bf16
// ---------------------------------------------------------------------------
typedef __attribute__((ext_vector_type(16))) __bf16 v16bf;
typedef __attribute__((ext_vector_type(8)))  float  v8f;

union Frag16 { v16bf v; unsigned u[8]; };

static __device__ __forceinline__ unsigned short f2bf(float f) {
  union { float f; unsigned u; } v; v.f = f;
  unsigned r = v.u + 0x7FFFu + ((v.u >> 16) & 1u);   // round-to-nearest-even
  return (unsigned short)(r >> 16);
}
static __device__ __forceinline__ unsigned pack2bf(float a, float b) {
  return (unsigned)f2bf(a) | ((unsigned)f2bf(b) << 16);
}
static __device__ __forceinline__ v8f wmma_bf16(v16bf a, v16bf b, v8f c) {
  return __builtin_amdgcn_wmma_f32_16x16x32_bf16(false, a, false, b, (short)0, c,
                                                 false, false);
}
// gfx1250 async global->LDS copy (ASYNCcnt-tracked); per-lane LDS dest address.
static __device__ __forceinline__ void async_copy_b128(unsigned lds_addr,
                                                       const void* gptr) {
  unsigned long long ga = (unsigned long long)gptr;
  asm volatile("global_load_async_to_lds_b128 %0, %1, off"
               :: "v"(lds_addr), "v"(ga) : "memory");
}
static __device__ __forceinline__ void wait_asynccnt0() {
  asm volatile("s_wait_asynccnt 0x0" ::: "memory");
}
static __device__ __forceinline__ unsigned lds_addr_of(const void* p) {
  return (unsigned)(unsigned long long)p;   // LDS aperture: addr[31:0] = LDS byte
}

// ---------------------------------------------------------------------------
// GEMM: out[M,N] = Abf[M,K](bf16) @ Wt[N,K](bf16, pre-transposed) + bias
// Block tile 128x128, 8 waves (wave tile 32x64), K-step 32.
// Double-buffered LDS with one-tile-ahead async prefetch (ASYNCcnt overlap).
// EPI: 0 = bias -> f32 out, 1 = bias + residual -> f32 out,
//      2 = bias + exact-erf GELU -> bf16 out
// ---------------------------------------------------------------------------
template<int EPI>
__global__ __launch_bounds__(256) void gemm_bf16_k(
    const unsigned short* __restrict__ Abf,
    const unsigned short* __restrict__ Wt,
    const float* __restrict__ bias, const float* __restrict__ res,
    float* __restrict__ outf, unsigned short* __restrict__ outbf,
    int M, int N, int K)
{
  __shared__ unsigned sA[2][128 * 16];  // [buf][row][kpair] (bf16x2 per u32)
  __shared__ unsigned sB[2][128 * 16];  // [buf][col][kpair]

  const int tid  = threadIdx.x;
  const int lane = tid & 31;
  const int wave = tid >> 5;
  const int hi   = lane >> 4;
  const int ln   = lane & 15;
  const int wm   = wave >> 1;     // 0..3
  const int wn   = wave & 1;      // 0..1
  const int row0 = blockIdx.y * 128;
  const int col0 = blockIdx.x * 128;

  // per-thread staging coordinates: 512 x 16B chunks per tile, 2 per thread
  const int sr  = tid >> 2;        // row/col 0..63 (first chunk set)
  const int sc4 = tid & 3;         // 16B chunk within 64B row

  auto issue_tile = [&](int k0, int buf) {
#pragma unroll
    for (int i = 0; i < 2; ++i) {
      int r = sr + i * 64;
      async_copy_b128(lds_addr_of(&sA[buf][r * 16 + sc4 * 4]),
                      Abf + (size_t)(row0 + r) * K + k0 + sc4 * 8);
      async_copy_b128(lds_addr_of(&sB[buf][r * 16 + sc4 * 4]),
                      Wt + (size_t)(col0 + r) * K + k0 + sc4 * 8);
    }
  };

  v8f acc[2][4];
#pragma unroll
  for (int mi = 0; mi < 2; ++mi)
#pragma unroll
    for (int ni = 0; ni < 4; ++ni)
#pragma unroll
      for (int e = 0; e < 8; ++e) acc[mi][ni][e] = 0.0f;

  const int T = K >> 5;
  issue_tile(0, 0);

  for (int t = 0; t < T; ++t) {
    const int buf = t & 1;
    wait_asynccnt0();      // copies issued last iteration (into buf) complete
    __syncthreads();       // visible to all; all waves done reading buf^1
    if (t + 1 < T) issue_tile((t + 1) << 5, buf ^ 1);  // prefetch next tile

    Frag16 af[2];
#pragma unroll
    for (int mi = 0; mi < 2; ++mi) {
      int r = wm * 32 + mi * 16 + ln;
#pragma unroll
      for (int p = 0; p < 8; ++p)
        af[mi].u[p] = sA[buf][r * 16 + ((p & 3) + hi * 4 + (p >> 2) * 8)];
    }
    Frag16 bf_[4];
#pragma unroll
    for (int ni = 0; ni < 4; ++ni) {
      int n = wn * 64 + ni * 16 + ln;
#pragma unroll
      for (int p = 0; p < 8; ++p)
        bf_[ni].u[p] = sB[buf][n * 16 + ((p & 3) + hi * 4 + (p >> 2) * 8)];
    }
#pragma unroll
    for (int mi = 0; mi < 2; ++mi)
#pragma unroll
      for (int ni = 0; ni < 4; ++ni)
        acc[mi][ni] = wmma_bf16(af[mi].v, bf_[ni].v, acc[mi][ni]);
  }

  // epilogue: C/D layout -> row = r + 8*hi, col = lane&15
#pragma unroll
  for (int mi = 0; mi < 2; ++mi)
#pragma unroll
    for (int ni = 0; ni < 4; ++ni)
#pragma unroll
      for (int r = 0; r < 8; ++r) {
        int row = row0 + wm * 32 + mi * 16 + r + hi * 8;
        int col = col0 + wn * 64 + ni * 16 + ln;
        float v = acc[mi][ni][r] + bias[col];
        if (EPI == 0) {
          outf[(size_t)row * N + col] = v;
        } else if (EPI == 1) {
          outf[(size_t)row * N + col] = v + res[(size_t)row * N + col];
        } else {
          v = 0.5f * v * (1.0f + erff(v * 0.70710678118654752f));
          outbf[(size_t)row * N + col] = f2bf(v);
        }
      }
}

// ---------------------------------------------------------------------------
// Flash attention: block = (b, h, 128 q-rows); 8 waves of 16 q-rows.
// q/k/v fp32 in, attn-out written as bf16 (feeds O-projection GEMM).
// ---------------------------------------------------------------------------
#define AT_S 1024
#define AT_D 512
#define AT_DH 64
#define AT_SCALE 0.125f

__global__ __launch_bounds__(256) void attn_k(
    const float* __restrict__ Q, const float* __restrict__ Kg,
    const float* __restrict__ Vg, unsigned short* __restrict__ Obf)
{
  __shared__ unsigned sK[64 * 32];          // [key][dpair]
  __shared__ unsigned sVT[64 * 32];         // [d][keypair]
  __shared__ unsigned short sP[8 * 16 * 64];// per-wave 16x64 bf16 P tile

  const int tid  = threadIdx.x;
  const int lane = tid & 31;
  const int wave = tid >> 5;
  const int hi   = lane >> 4;
  const int ln   = lane & 15;
  const int qt = blockIdx.x, h = blockIdx.y, b = blockIdx.z;
  const int q0 = qt * 128 + wave * 16;
  const int wb = wave * (16 * 64);

  Frag16 aq[2];
  {
    const float* qp = Q + ((size_t)(b * AT_S + q0 + ln)) * AT_D + h * AT_DH;
#pragma unroll
    for (int s = 0; s < 2; ++s)
#pragma unroll
      for (int p = 0; p < 8; ++p) {
        int k = s * 32 + (p & 3) * 2 + hi * 8 + (p >> 2) * 16;
        aq[s].u[p] = pack2bf(qp[k], qp[k + 1]);
      }
  }

  v8f o[4];
#pragma unroll
  for (int n = 0; n < 4; ++n)
#pragma unroll
    for (int e = 0; e < 8; ++e) o[n][e] = 0.0f;
  float m[8], l[8];
#pragma unroll
  for (int r = 0; r < 8; ++r) { m[r] = -1e30f; l[r] = 0.0f; }

  for (int kt = 0; kt < AT_S / 64; ++kt) {
    __syncthreads();
#pragma unroll
    for (int i = 0; i < 8; ++i) {
      int pi  = tid + i * 256;
      int key = pi >> 5;
      int dp  = pi & 31;
      size_t base = ((size_t)(b * AT_S + kt * 64 + key)) * AT_D + h * AT_DH + dp * 2;
      sK[key * 32 + dp] = pack2bf(Kg[base], Kg[base + 1]);
      unsigned short v0 = f2bf(Vg[base]), v1 = f2bf(Vg[base + 1]);
      ((unsigned short*)sVT)[(dp * 2)     * 64 + key] = v0;
      ((unsigned short*)sVT)[(dp * 2 + 1) * 64 + key] = v1;
    }
    __syncthreads();

    v8f s4[4];
#pragma unroll
    for (int n = 0; n < 4; ++n) {
      Frag16 bk0, bk1;
      int key = n * 16 + ln;
#pragma unroll
      for (int p = 0; p < 8; ++p) {
        bk0.u[p] = sK[key * 32 + ((p & 3) + hi * 4 + (p >> 2) * 8)];
        bk1.u[p] = sK[key * 32 + 16 + ((p & 3) + hi * 4 + (p >> 2) * 8)];
      }
      v8f z;
#pragma unroll
      for (int e = 0; e < 8; ++e) z[e] = 0.0f;
      z = wmma_bf16(aq[0].v, bk0.v, z);
      z = wmma_bf16(aq[1].v, bk1.v, z);
      s4[n] = z;
    }

    float alpha[8], rs[8];
#pragma unroll
    for (int r = 0; r < 8; ++r) {
      float vmax = fmaxf(fmaxf(s4[0][r], s4[1][r]), fmaxf(s4[2][r], s4[3][r]));
#pragma unroll
      for (int off = 1; off < 16; off <<= 1)
        vmax = fmaxf(vmax, __shfl_xor(vmax, off, 32));
      float mnew = fmaxf(m[r], vmax * AT_SCALE);
      alpha[r] = __expf(m[r] - mnew);
      m[r] = mnew;
      rs[r] = 0.0f;
    }
#pragma unroll
    for (int n = 0; n < 4; ++n)
#pragma unroll
      for (int r = 0; r < 8; ++r) {
        float pv = __expf(s4[n][r] * AT_SCALE - m[r]);
        rs[r] += pv;
        sP[wb + (r + hi * 8) * 64 + n * 16 + ln] = f2bf(pv);
      }
#pragma unroll
    for (int r = 0; r < 8; ++r) {
      float s = rs[r];
#pragma unroll
      for (int off = 1; off < 16; off <<= 1) s += __shfl_xor(s, off, 32);
      l[r] = l[r] * alpha[r] + s;
#pragma unroll
      for (int n = 0; n < 4; ++n) o[n][r] *= alpha[r];
    }
    __syncthreads();

    Frag16 pa[2];
#pragma unroll
    for (int s = 0; s < 2; ++s)
#pragma unroll
      for (int p = 0; p < 8; ++p) {
        int kk = s * 32 + (p & 3) * 2 + hi * 8 + (p >> 2) * 16;
        pa[s].u[p] = *(const unsigned*)&sP[wb + ln * 64 + kk];
      }
#pragma unroll
    for (int n = 0; n < 4; ++n) {
      Frag16 bv0, bv1;
      int d = n * 16 + ln;
#pragma unroll
      for (int p = 0; p < 8; ++p) {
        bv0.u[p] = sVT[d * 32 + ((p & 3) + hi * 4 + (p >> 2) * 8)];
        bv1.u[p] = sVT[d * 32 + 16 + ((p & 3) + hi * 4 + (p >> 2) * 8)];
      }
      o[n] = wmma_bf16(pa[0].v, bv0.v, o[n]);
      o[n] = wmma_bf16(pa[1].v, bv1.v, o[n]);
    }
  }

#pragma unroll
  for (int n = 0; n < 4; ++n)
#pragma unroll
    for (int r = 0; r < 8; ++r) {
      int row = q0 + r + hi * 8;
      int d   = h * AT_DH + n * 16 + ln;
      Obf[((size_t)(b * AT_S + row)) * AT_D + d] = f2bf(o[n][r] / l[r]);
    }
}

// ---------------------------------------------------------------------------
// LayerNorm over D=512: writes fp32 X and bf16 copy (next GEMM's A operand)
// ---------------------------------------------------------------------------
__global__ __launch_bounds__(256) void layernorm_k(
    const float* __restrict__ y, const float* __restrict__ g,
    const float* __restrict__ be, float* __restrict__ out,
    unsigned short* __restrict__ outbf)
{
  const int row = blockIdx.x;
  const int tid = threadIdx.x;
  const float* p = y + (size_t)row * 512;
  float a0 = p[tid], a1 = p[tid + 256];
  float s = a0 + a1, q = a0 * a0 + a1 * a1;
#pragma unroll
  for (int off = 1; off < 32; off <<= 1) {
    s += __shfl_xor(s, off, 32);
    q += __shfl_xor(q, off, 32);
  }
  __shared__ float rs[8], rq[8], mv[2];
  if ((tid & 31) == 0) { rs[tid >> 5] = s; rq[tid >> 5] = q; }
  __syncthreads();
  if (tid == 0) {
    float S = 0.f, Qs = 0.f;
    for (int i = 0; i < 8; ++i) { S += rs[i]; Qs += rq[i]; }
    float mean = S * (1.0f / 512.0f);
    float var  = Qs * (1.0f / 512.0f) - mean * mean;
    mv[0] = mean; mv[1] = rsqrtf(var + 1e-5f);
  }
  __syncthreads();
  float mean = mv[0], rstd = mv[1];
  float o0 = (a0 - mean) * rstd * g[tid]       + be[tid];
  float o1 = (a1 - mean) * rstd * g[tid + 256] + be[tid + 256];
  out[(size_t)row * 512 + tid]       = o0;
  out[(size_t)row * 512 + tid + 256] = o1;
  outbf[(size_t)row * 512 + tid]       = f2bf(o0);
  outbf[(size_t)row * 512 + tid + 256] = f2bf(o1);
}

// ---------------------------------------------------------------------------
// Weight convert + transpose: W[K,N] fp32 -> Wt[N,K] bf16
// ---------------------------------------------------------------------------
__global__ __launch_bounds__(256) void wt_k(const float* __restrict__ W,
                                            unsigned short* __restrict__ Wt,
                                            int K, int N) {
  int idx = blockIdx.x * 256 + threadIdx.x;
  int n = idx % N, k = idx / N;
  Wt[(size_t)n * K + k] = f2bf(W[idx]);
}

// fp32 -> bf16 elementwise (initial X conversion)
__global__ __launch_bounds__(256) void cvt_k(const float* __restrict__ in,
                                             unsigned short* __restrict__ out) {
  int idx = blockIdx.x * 256 + threadIdx.x;
  out[idx] = f2bf(in[idx]);
}

// ---------------------------------------------------------------------------
// Host orchestration
// ---------------------------------------------------------------------------
extern "C" void kernel_launch(void* const* d_in, const int* in_sizes, int n_in,
                              void* d_out, int out_size, void* d_ws, size_t ws_size,
                              hipStream_t stream) {
  const int L = 8, D = 512, F = 2048, M = 4096;  // M = B*S
  const float* x  = (const float*)d_in[0];
  const float* wq = (const float*)d_in[1];  const float* bq = (const float*)d_in[2];
  const float* wk = (const float*)d_in[3];  const float* bk = (const float*)d_in[4];
  const float* wv = (const float*)d_in[5];  const float* bv = (const float*)d_in[6];
  const float* wo = (const float*)d_in[7];  const float* bo = (const float*)d_in[8];
  const float* g1 = (const float*)d_in[9];  const float* be1 = (const float*)d_in[10];
  const float* w1 = (const float*)d_in[11]; const float* b1  = (const float*)d_in[12];
  const float* w2 = (const float*)d_in[13]; const float* b2  = (const float*)d_in[14];
  const float* g2 = (const float*)d_in[15]; const float* be2 = (const float*)d_in[16];

  float* X = (float*)d_out;                            // running activations
  char* w8 = (char*)d_ws;
  float* qf = (float*)w8;                              // [M,D] f32
  float* kf = qf + (size_t)M * D;
  float* vf = kf + (size_t)M * D;
  unsigned short* Xbf  = (unsigned short*)(vf + (size_t)M * D); // [M,D] bf16
  unsigned short* t0bf = Xbf  + (size_t)M * D;                  // [M,D] bf16
  unsigned short* hbf  = t0bf + (size_t)M * D;                  // [M,F] bf16
  unsigned short* Wqt  = hbf  + (size_t)M * F;                  // per-layer weights
  unsigned short* Wkt  = Wqt + (size_t)D * D;
  unsigned short* Wvt  = Wkt + (size_t)D * D;
  unsigned short* Wot  = Wvt + (size_t)D * D;
  unsigned short* W1t  = Wot + (size_t)D * D;                   // [F,D]
  unsigned short* W2t  = W1t + (size_t)D * F;                   // [D,F]

  hipMemcpyAsync(X, x, (size_t)M * D * sizeof(float),
                 hipMemcpyDeviceToDevice, stream);

  dim3 blk(256);
  dim3 gD(D / 128, M / 128);
  dim3 gF(F / 128, M / 128);
  dim3 gA(8, 8, 4);
  const int gDD = D * D / 256, gDF = D * F / 256;

  cvt_k<<<dim3(M * D / 256), blk, 0, stream>>>(X, Xbf);

  for (int l = 0; l < L; ++l) {
    const float* Wq = wq + (size_t)l * D * D;  const float* Bq = bq + (size_t)l * D;
    const float* Wk = wk + (size_t)l * D * D;  const float* Bk = bk + (size_t)l * D;
    const float* Wv = wv + (size_t)l * D * D;  const float* Bv = bv + (size_t)l * D;
    const float* Wo = wo + (size_t)l * D * D;  const float* Bo = bo + (size_t)l * D;
    const float* W1 = w1 + (size_t)l * D * F;  const float* B1 = b1 + (size_t)l * F;
    const float* W2 = w2 + (size_t)l * F * D;  const float* B2 = b2 + (size_t)l * D;
    const float* G1 = g1 + (size_t)l * D;      const float* E1 = be1 + (size_t)l * D;
    const float* G2 = g2 + (size_t)l * D;      const float* E2 = be2 + (size_t)l * D;

    wt_k<<<dim3(gDD), blk, 0, stream>>>(Wq, Wqt, D, D);
    wt_k<<<dim3(gDD), blk, 0, stream>>>(Wk, Wkt, D, D);
    wt_k<<<dim3(gDD), blk, 0, stream>>>(Wv, Wvt, D, D);
    wt_k<<<dim3(gDD), blk, 0, stream>>>(Wo, Wot, D, D);
    wt_k<<<dim3(gDF), blk, 0, stream>>>(W1, W1t, D, F);
    wt_k<<<dim3(gDF), blk, 0, stream>>>(W2, W2t, F, D);

    gemm_bf16_k<0><<<gD, blk, 0, stream>>>(Xbf, Wqt, Bq, nullptr, qf, nullptr, M, D, D);
    gemm_bf16_k<0><<<gD, blk, 0, stream>>>(Xbf, Wkt, Bk, nullptr, kf, nullptr, M, D, D);
    gemm_bf16_k<0><<<gD, blk, 0, stream>>>(Xbf, Wvt, Bv, nullptr, vf, nullptr, M, D, D);
    attn_k<<<gA, blk, 0, stream>>>(qf, kf, vf, t0bf);
    gemm_bf16_k<1><<<gD, blk, 0, stream>>>(t0bf, Wot, Bo, X, qf, nullptr, M, D, D);
    layernorm_k<<<dim3(M), blk, 0, stream>>>(qf, G1, E1, X, Xbf);
    gemm_bf16_k<2><<<gF, blk, 0, stream>>>(Xbf, W1t, B1, nullptr, nullptr, hbf, M, F, D);
    gemm_bf16_k<1><<<gD, blk, 0, stream>>>(hbf, W2t, B2, X, qf, nullptr, M, D, F);
    layernorm_k<<<dim3(M), blk, 0, stream>>>(qf, G2, E2, X, Xbf);
  }
}
